// pLoss_7017976562214
// MI455X (gfx1250) — matrix operand order repaired and checked by reading, not compiled, not run on hardware
//
#include <hip/hip_runtime.h>

// ---------------------------------------------------------------------------
// pLoss fused kernel for MI455X (gfx1250, wave32, WMMA).
//
//   potential = S (32768x64) x f^T (64x512)      -> 16x16 tiles in VGPRs
//   P         = exp(potential) / Z  (softmax over states; |pot| <~ 40 so no
//               max-subtraction needed in f32/bf16-range arithmetic)
//   pMargin   = P^T x S                          -> fused: exp'd tile feeds
//                                                   directly as WMMA A operand
//   loss      = masked BCE, mean over batch, sum over labels
//
// Key point: the 64MB intermediate `potential` never touches memory; each
// 16x16 f32 D-tile of GEMM1 is exponentiated per-lane, packed to bf16, and is
// *bit-layout-identical* to the A operand of GEMM2 (D: lane n = batch col n,
// vgpr r = state r / 16+r  ==  A: lane n = batch row n, K = state). No shuffles.
// ---------------------------------------------------------------------------

typedef __attribute__((ext_vector_type(16))) __bf16 v16bf;
typedef __attribute__((ext_vector_type(8)))  __bf16 v8bf;
typedef __attribute__((ext_vector_type(8)))  float  v8f;

#define WMMA_BF16(A, B, C) \
  __builtin_amdgcn_wmma_f32_16x16x32_bf16(false, (A), false, (B), (short)0, (C), false, false)

constexpr int NUM_STATES = 32768;
constexpr int NUM_LABELS = 64;
constexpr int BATCH      = 512;
constexpr int SPLIT      = 32;                 // state chunks
constexpr int CHUNK      = NUM_STATES / SPLIT; // 1024 states per chunk

// ---------------- workspace layout (bytes) ----------------
constexpr size_t WS_FBF   = 0;                                  // 512*64 bf16   = 64KB
constexpr size_t WS_SBF   = WS_FBF  + (size_t)BATCH * 64 * 2;   // 32768*64 bf16 = 4MB
constexpr size_t WS_STBF  = WS_SBF  + (size_t)NUM_STATES * 64 * 2;  // 64*32768 bf16 = 4MB
constexpr size_t WS_ACCP  = WS_STBF + (size_t)64 * NUM_STATES * 2;  // SPLIT*512*64 f32 = 4MB
constexpr size_t WS_ZP    = WS_ACCP + (size_t)SPLIT * BATCH * 64 * 4; // SPLIT*512 f32
constexpr size_t WS_LOSSP = WS_ZP   + (size_t)SPLIT * BATCH * 4;      // 128 f32
// total ~12.8 MB

// ---------------------------------------------------------------------------
// Prep: convert f -> bf16, S -> bf16 (row major) and bf16 transposed.
// ---------------------------------------------------------------------------
__global__ __launch_bounds__(256)
void k_prep(const float* __restrict__ f, const float* __restrict__ S,
            __bf16* __restrict__ fbf, __bf16* __restrict__ Sbf,
            __bf16* __restrict__ Stbf) {
  int idx = blockIdx.x * 256 + threadIdx.x;   // 0 .. 32768*64-1
  float sv = S[idx];
  __bf16 sb = (__bf16)sv;
  Sbf[idx] = sb;
  int row = idx >> 6;        // state
  int col = idx & 63;        // label
  Stbf[col * NUM_STATES + row] = sb;
  if (idx < BATCH * NUM_LABELS) fbf[idx] = (__bf16)f[idx];
}

// ---------------------------------------------------------------------------
// Fused flash kernel: grid (32 batch tiles, SPLIT chunks), 4 waves per WG.
// Per 32-state step: 4 WMMA (S x f^T) + exp + 4 WMMA (P^T x S).
// ---------------------------------------------------------------------------
__global__ __launch_bounds__(128)
void k_fused(const __bf16* __restrict__ fbf, const __bf16* __restrict__ Sbf,
             const __bf16* __restrict__ Stbf,
             float* __restrict__ accP, float* __restrict__ ZP) {
  const int bt    = blockIdx.x * 16;     // batch tile base
  const int chunk = blockIdx.y;
  const int wave  = threadIdx.x >> 5;
  const int lane  = threadIdx.x & 31;
  const int n     = lane & 15;           // column index within tile
  const int half  = lane >> 4;           // 0: lanes 0-15, 1: lanes 16-31

  // GEMM1 B operand (f^T tiles), constant over the whole state loop.
  // B layout: lane(n,half) holds K = half*16 + {0..15} of column n  -> f is
  // row-major [batch][label], so this is a contiguous 32B load per lane.
  const __bf16* fb = fbf + (bt + n) * 64 + half * 16;
  const v16bf Bf0 = *(const v16bf*)(fb);        // labels 0..31
  const v16bf Bf1 = *(const v16bf*)(fb + 32);   // labels 32..63

  v8f acc0 = {}, acc1 = {}, acc2 = {}, acc3 = {};
  float Z = 0.0f;

  const int sEnd = chunk * CHUNK + CHUNK;
  for (int s = chunk * CHUNK + wave * 32; s < sEnd; s += 128) {
    // ---- GEMM1: potential tiles for states s+{0..7,16..23} and s+{8..15,24..31} ----
    // A layout: lane(n,half): elems 0..15 = K = kbase+half*16+{0..15}
    v8f pot0 = {}, pot1 = {};
    {
      const __bf16* ap = Sbf + (size_t)(s + n + (n >> 3) * 8) * 64 + half * 16;
      v8bf lo, hi; v16bf A;

      lo = *(const v8bf*)(ap);       hi = *(const v8bf*)(ap + 8);
      A  = __builtin_shufflevector(lo, hi, 0,1,2,3,4,5,6,7,8,9,10,11,12,13,14,15);
      pot0 = WMMA_BF16(A, Bf0, pot0);
      lo = *(const v8bf*)(ap + 32);  hi = *(const v8bf*)(ap + 40);
      A  = __builtin_shufflevector(lo, hi, 0,1,2,3,4,5,6,7,8,9,10,11,12,13,14,15);
      pot0 = WMMA_BF16(A, Bf1, pot0);

      ap += 8 * 64;                  // states s+8..s+15, s+24..s+31
      lo = *(const v8bf*)(ap);       hi = *(const v8bf*)(ap + 8);
      A  = __builtin_shufflevector(lo, hi, 0,1,2,3,4,5,6,7,8,9,10,11,12,13,14,15);
      pot1 = WMMA_BF16(A, Bf0, pot1);
      lo = *(const v8bf*)(ap + 32);  hi = *(const v8bf*)(ap + 40);
      A  = __builtin_shufflevector(lo, hi, 0,1,2,3,4,5,6,7,8,9,10,11,12,13,14,15);
      pot1 = WMMA_BF16(A, Bf1, pot1);
    }

    // ---- exp + pack: D tile (lane n = batch col n, vgpr r = state) is the
    //      exact A layout for GEMM2 (lane n = batch row n, K = state). ----
    v16bf A2;
    float zs = 0.0f;
#pragma unroll
    for (int r = 0; r < 8; ++r) {
      float e0 = __expf(pot0[r]);    // states K = half*16 + r      (of 32)
      float e1 = __expf(pot1[r]);    // states K = half*16 + 8 + r
      zs += e0 + e1;
      A2[r]     = (__bf16)e0;
      A2[r + 8] = (__bf16)e1;
    }
    Z += zs;

    // ---- GEMM2: acc[batch, label] += P^T x S over these 32 states ----
    // B layout: lane(n,half) holds K = half*16+{0..15} of label column n
    //   -> contiguous 32B per lane from the transposed S.
    {
      const __bf16* stp = Stbf + (size_t)n * NUM_STATES + s + half * 16;
      v16bf B2;
      B2 = *(const v16bf*)(stp);                    acc0 = WMMA_BF16(A2, B2, acc0);
      B2 = *(const v16bf*)(stp + 16 * NUM_STATES);  acc1 = WMMA_BF16(A2, B2, acc1);
      B2 = *(const v16bf*)(stp + 32 * NUM_STATES);  acc2 = WMMA_BF16(A2, B2, acc2);
      B2 = *(const v16bf*)(stp + 48 * NUM_STATES);  acc3 = WMMA_BF16(A2, B2, acc3);
    }
  }

  // ---- combine the 4 waves deterministically through LDS ----
  __shared__ float accL[4][16][64];   // [wave][batch row][label]
  __shared__ float zL[4][32];
#pragma unroll
  for (int r = 0; r < 8; ++r) {
    // acc D layout: lane(n,half): vgpr r = [batch row half*8+r][label tile col n]
    accL[wave][half * 8 + r][n]      = acc0[r];
    accL[wave][half * 8 + r][16 + n] = acc1[r];
    accL[wave][half * 8 + r][32 + n] = acc2[r];
    accL[wave][half * 8 + r][48 + n] = acc3[r];
  }
  zL[wave][lane] = Z;   // lane covers batch column n (both halves contribute)
  __syncthreads();

  for (int o = threadIdx.x; o < 16 * 64; o += 128) {
    int br = o >> 6, lb = o & 63;
    float v = accL[0][br][lb] + accL[1][br][lb] + accL[2][br][lb] + accL[3][br][lb];
    accP[((size_t)chunk * BATCH + bt + br) * 64 + lb] = v;
  }
  if (threadIdx.x < 16) {
    float zz = 0.0f;
#pragma unroll
    for (int w = 0; w < 4; ++w) zz += zL[w][threadIdx.x] + zL[w][threadIdx.x + 16];
    ZP[chunk * BATCH + bt + threadIdx.x] = zz;
  }
}

// ---------------------------------------------------------------------------
// Reduce chunk partials, normalize, write pMargin, masked BCE partial sums.
// ---------------------------------------------------------------------------
__global__ __launch_bounds__(256)
void k_final(const float* __restrict__ accP, const float* __restrict__ ZP,
             const float* __restrict__ y, const float* __restrict__ mask,
             float* __restrict__ out, float* __restrict__ lossP) {
  int idx = blockIdx.x * 256 + threadIdx.x;   // 0 .. 512*64-1
  int b = idx >> 6, i = idx & 63;
  float zsum = 0.0f, asum = 0.0f;
#pragma unroll
  for (int c = 0; c < SPLIT; ++c) {
    zsum += ZP[c * BATCH + b];
    asum += accP[((size_t)c * BATCH + b) * 64 + i];
  }
  float pM = asum / zsum;
  out[1 + idx] = pM;

  float lp  = fmaxf(__logf(pM), -100.0f);
  float l1p = fmaxf(__logf(1.0f - pM), -100.0f);
  float yv = y[idx];
  float bce = -(yv * lp + (1.0f - yv) * l1p) * mask[idx];

  __shared__ float red[256];
  red[threadIdx.x] = bce * (1.0f / (float)BATCH);
  __syncthreads();
#pragma unroll
  for (int off = 128; off > 0; off >>= 1) {
    if (threadIdx.x < off) red[threadIdx.x] += red[threadIdx.x + off];
    __syncthreads();
  }
  if (threadIdx.x == 0) lossP[blockIdx.x] = red[0];
}

// ---------------------------------------------------------------------------
// Final deterministic loss sum (128 block partials -> out[0]).
// ---------------------------------------------------------------------------
__global__ __launch_bounds__(128)
void k_loss(const float* __restrict__ lossP, float* __restrict__ out) {
  __shared__ float red[128];
  red[threadIdx.x] = lossP[threadIdx.x];
  __syncthreads();
#pragma unroll
  for (int off = 64; off > 0; off >>= 1) {
    if (threadIdx.x < off) red[threadIdx.x] += red[threadIdx.x + off];
    __syncthreads();
  }
  if (threadIdx.x == 0) out[0] = red[0];
}

// ---------------------------------------------------------------------------
extern "C" void kernel_launch(void* const* d_in, const int* in_sizes, int n_in,
                              void* d_out, int out_size, void* d_ws, size_t ws_size,
                              hipStream_t stream) {
  const float* f    = (const float*)d_in[0];   // [512,64]
  const float* S    = (const float*)d_in[1];   // [32768,64]
  const float* y    = (const float*)d_in[2];   // [512,64]
  const float* mask = (const float*)d_in[3];   // [512,64]
  float* out = (float*)d_out;                  // [1 + 512*64]

  char* ws = (char*)d_ws;                      // needs ~12.8 MB
  __bf16* fbf   = (__bf16*)(ws + WS_FBF);
  __bf16* Sbf   = (__bf16*)(ws + WS_SBF);
  __bf16* Stbf  = (__bf16*)(ws + WS_STBF);
  float*  accP  = (float*)(ws + WS_ACCP);
  float*  ZP    = (float*)(ws + WS_ZP);
  float*  lossP = (float*)(ws + WS_LOSSP);

  k_prep <<<(NUM_STATES * NUM_LABELS) / 256, 256, 0, stream>>>(f, S, fbf, Sbf, Stbf);
  k_fused<<<dim3(BATCH / 16, SPLIT), 128, 0, stream>>>(fbf, Sbf, Stbf, accP, ZP);
  k_final<<<(BATCH * NUM_LABELS) / 256, 256, 0, stream>>>(accP, ZP, y, mask, out, lossP);
  k_loss <<<1, 128, 0, stream>>>(lossP, out);
}
